// SparseEmbed1d_24008867184743
// MI455X (gfx1250) — matrix operand order repaired and checked by reading, not compiled, not run on hardware
//
#include <hip/hip_runtime.h>

#define BATCH 8192
#define VOCAB 16384
#define EMBED 256
#define BM    64
#define KT    32
#define ITERS (VOCAB / KT)
#define EPS   1e-6f

// LDS B tile: 32 rows x (512B data + 16B pad) -> bank-conflict-friendly pitch.
// 3-buffer ring: 2-deep TDM pipeline + single barrier per iteration.
#define BROW_BYTES  528
#define BTILE_BYTES (KT * BROW_BYTES)   // 16896 B
#define NBUF        3

typedef __attribute__((ext_vector_type(16))) _Float16     v16h;
typedef __attribute__((ext_vector_type(8)))  _Float16     v8h;
typedef __attribute__((ext_vector_type(4)))  _Float16     v4h;
typedef __attribute__((ext_vector_type(8)))  float        v8f;
typedef __attribute__((ext_vector_type(4)))  unsigned int u32x4;
typedef __attribute__((ext_vector_type(8)))  int          i32x8;
typedef __attribute__((ext_vector_type(4)))  int          i32x4;

// ---------------------------------------------------------------------------
// TDM: load a 32-row x 256-half tile of emb_f16 (row-major, pitch 256 halfs)
// into LDS at lds_off, inserting 4 dwords of pad after each 512-byte row.
// ---------------------------------------------------------------------------
__device__ __forceinline__ void tdm_load_b(const _Float16* gsrc, unsigned lds_off) {
  unsigned long long ga = (unsigned long long)(uintptr_t)gsrc;
  u32x4 g0;
  g0[0] = 1u;                                        // count=1, user mode
  g0[1] = lds_off;                                   // LDS byte address
  g0[2] = (unsigned)(ga & 0xFFFFFFFFu);              // global_addr[31:0]
  g0[3] = (unsigned)((ga >> 32) & 0x1FFFFFFu)        // global_addr[56:32]
        | (2u << 30);                                // type=2 ("image")
  i32x8 g1;
  g1[0] = (int)((1u << 16)      // data_size = 2 bytes
              | (1u << 20)      // pad_enable
              | (6u << 22)      // pad_interval: 128 dwords (512B row)
              | (3u << 25));    // pad_amount: 4 dwords (16B)
  g1[1] = (int)(256u   << 16);  // tensor_dim0 = 256   (bits 79:48, low part)
  g1[2] = (int)(16384u << 16);  // dim0 hi=0 | tensor_dim1 = 16384 (low 16)
  g1[3] = (int)(256u   << 16);  // dim1 hi=0 | tile_dim0 = 256
  g1[4] = (int)KT;              // tile_dim1 = 32, tile_dim2 = 0
  g1[5] = 256;                  // tensor_dim0_stride = 256 (low 32)
  g1[6] = 0;                    // stride0 hi | tensor_dim1_stride lo
  g1[7] = 0;
  i32x4 gz = {0, 0, 0, 0};
#if __clang_major__ >= 23
  i32x8 gz8 = {0, 0, 0, 0, 0, 0, 0, 0};
  __builtin_amdgcn_tensor_load_to_lds(g0, g1, gz, gz, gz8, 0);
#else
  __builtin_amdgcn_tensor_load_to_lds(g0, g1, gz, gz, 0);
#endif
}

// ---------------------------------------------------------------------------
// Kernel 1: emb f32 -> f16 into workspace (read 16MB, write 8MB; negligible)
// ---------------------------------------------------------------------------
__global__ __launch_bounds__(256) void cvt_emb_f16(const float* __restrict__ e,
                                                   _Float16* __restrict__ o) {
  size_t i = ((size_t)blockIdx.x * blockDim.x + threadIdx.x) * 4;
  float4 f = *(const float4*)(e + i);
  v4h h = { (_Float16)f.x, (_Float16)f.y, (_Float16)f.z, (_Float16)f.w };
  *(v4h*)(o + i) = h;
}

// x values are {0,1}: packed f16 pair = (lo + (hi<<16)) * 0x3C00 (2 VALU ops)
#define PACK01(lo, hi) ((unsigned)((lo) + ((hi) << 16)) * 0x3C00u)

// ---------------------------------------------------------------------------
// Kernel 2: WMMA GEMM  out = (x!=0) @ emb_f16 / (rowcount + eps)
// Grid: 128 WGs of 256 threads. WG covers 64 rows x 256 cols (x read once).
// Waves: 4 M-tiles x 2 N-halves; wave owns 8 accumulator tiles of 16x16 f32.
// ---------------------------------------------------------------------------
__global__ __launch_bounds__(256) void sparse_embed_gemm(
    const int* __restrict__ x, const _Float16* __restrict__ embh,
    float* __restrict__ out)
{
  __shared__ __align__(128) unsigned char bsm[NBUF * BTILE_BYTES];
  __shared__ int cnts[BM];

  const int tid    = threadIdx.x;
  const int lane   = tid & 31;
  const int wid    = tid >> 5;     // 0..7
  const int wave_m = wid & 3;      // which 16-row M tile
  const int wave_n = wid >> 2;     // which 128-col N half
  const int m      = lane & 15;
  const int halfid = lane >> 4;

  const int rowblk = blockIdx.x * BM;
  const long long arow = (long long)(rowblk + wave_m * 16 + m);
  const int* __restrict__ xrow = x + arow * (long long)VOCAB;

  v8f acc[8];
#pragma unroll
  for (int j = 0; j < 8; ++j) acc[j] = (v8f){0.f,0.f,0.f,0.f,0.f,0.f,0.f,0.f};

  // Prologue: 2-deep TDM pipeline (tiles 0 and 1), raw A for iteration 0.
  if (wid == 0) {
    tdm_load_b(embh, (unsigned)(uintptr_t)&bsm[0]);
    tdm_load_b(embh + (size_t)KT * EMBED,
               (unsigned)(uintptr_t)&bsm[BTILE_BYTES]);
  }

  int4 ra[2][4];
  {
    const int4* p0 = (const int4*)(xrow + halfid * 8);
    const int4* p1 = (const int4*)(xrow + 16 + halfid * 8);
    ra[0][0] = p0[0]; ra[0][1] = p0[1]; ra[0][2] = p1[0]; ra[0][3] = p1[1];
  }

  int cnt  = 0;
  int bcur = 0;                    // buffer holding tile `it`
  int biss = 2;                    // buffer for tile `it+2` (ring of 3)

#pragma unroll 2
  for (int it = 0; it < ITERS; ++it) {
    const int k0 = it * KT;
    const int cb = it & 1, nb = cb ^ 1;

    // ---- issue raw A loads for next iteration (clamped; redundant on last)
    {
      const int k1 = (it + 1 < ITERS) ? k0 + KT : k0;
      const int4* p0 = (const int4*)(xrow + k1 + halfid * 8);
      const int4* p1 = (const int4*)(xrow + k1 + 16 + halfid * 8);
      ra[nb][0] = p0[0]; ra[nb][1] = p0[1]; ra[nb][2] = p1[0]; ra[nb][3] = p1[1];
      __builtin_prefetch(xrow + k0 + 2048 + halfid * 8, 0, 1);
    }

    // ---- count + convert current A (lanes 0-15: K 0-7,16-23; 16-31: 8-15,24-31)
    int4 a0 = ra[cb][0], a1 = ra[cb][1], a2 = ra[cb][2], a3 = ra[cb][3];
    if (wave_n == 0)   // count each (row,k) exactly once; x values are {0,1}
      cnt += a0.x + a0.y + a0.z + a0.w + a1.x + a1.y + a1.z + a1.w
           + a2.x + a2.y + a2.z + a2.w + a3.x + a3.y + a3.z + a3.w;

    union { v16h v; unsigned u[8]; } af;
    af.u[0] = PACK01(a0.x, a0.y);  af.u[1] = PACK01(a0.z, a0.w);
    af.u[2] = PACK01(a1.x, a1.y);  af.u[3] = PACK01(a1.z, a1.w);
    af.u[4] = PACK01(a2.x, a2.y);  af.u[5] = PACK01(a2.z, a2.w);
    af.u[6] = PACK01(a3.x, a3.y);  af.u[7] = PACK01(a3.z, a3.w);

    // ---- tile `it` landed? (TDM completes in order; <=2 in flight)
    if (wid == 0) {
      if (it + 1 < ITERS) __builtin_amdgcn_s_wait_tensorcnt(1);
      else                __builtin_amdgcn_s_wait_tensorcnt(0);
    }
    __syncthreads();   // all waves done with compute(it-1); tile `it` visible

    // ---- refill the ring: tile it+2 -> buffer that compute(it-1) just freed
    if (wid == 0 && it + 2 < ITERS)
      tdm_load_b(embh + (size_t)(k0 + 2 * KT) * EMBED,
                 (unsigned)(uintptr_t)&bsm[biss * BTILE_BYTES]);

    // ---- preload all 8 B fragments (one ds clause, one wait), then 8 WMMAs
    const unsigned char* bbase = &bsm[bcur * BTILE_BYTES]
                               + (unsigned)(8 * halfid) * BROW_BYTES
                               + (unsigned)wave_n * 256   // 128 cols * 2B
                               + (unsigned)m * 2;
    union { v16h v; v8h h[2]; } bf[8];
#pragma unroll
    for (int j = 0; j < 8; ++j) {
#pragma unroll
      for (int s = 0; s < 16; ++s) {
        const int kk = (s < 8) ? s : s + 8;
        bf[j].v[s] = *(const _Float16*)(bbase + kk * BROW_BYTES + j * 32);
      }
    }
#pragma unroll
    for (int j = 0; j < 8; ++j)
      acc[j] = __builtin_amdgcn_wmma_f32_16x16x32_f16(
          false, af.v, false, bf[j].v, (short)0, acc[j], false, false);

    bcur = (bcur == NBUF - 1) ? 0 : bcur + 1;
    biss = (biss == NBUF - 1) ? 0 : biss + 1;
  }

  // ---- per-row counts: lane m + lane m+16 hold complementary K subsets
  if (wave_n == 0) {
    int tot = cnt + __shfl_xor(cnt, 16, 32);
    if (lane < 16) cnts[wave_m * 16 + lane] = tot;
  }
  __syncthreads();

  // ---- epilogue: scale by 1/(count+eps) and store (C layout: VGPR r -> M=r|r+8)
  float inv[8];
#pragma unroll
  for (int r = 0; r < 8; ++r) {
    int rl = wave_m * 16 + r + 8 * halfid;
    inv[r] = 1.0f / ((float)cnts[rl] + EPS);
  }
#pragma unroll
  for (int j = 0; j < 8; ++j) {
    const int ncol = wave_n * 128 + j * 16 + m;
#pragma unroll
    for (int r = 0; r < 8; ++r) {
      const int grow = rowblk + wave_m * 16 + r + 8 * halfid;
      out[(size_t)grow * EMBED + ncol] = acc[j][r] * inv[r];
    }
  }
}

// ---------------------------------------------------------------------------
extern "C" void kernel_launch(void* const* d_in, const int* in_sizes, int n_in,
                              void* d_out, int out_size, void* d_ws, size_t ws_size,
                              hipStream_t stream) {
  (void)in_sizes; (void)n_in; (void)out_size; (void)ws_size;
  const int*   x   = (const int*)d_in[0];
  const float* emb = (const float*)d_in[1];
  float*       out = (float*)d_out;
  _Float16*    embh = (_Float16*)d_ws;   // 8 MB scratch: emb in f16

  cvt_emb_f16<<<(VOCAB * EMBED) / (256 * 4), 256, 0, stream>>>(emb, embh);
  sparse_embed_gemm<<<BATCH / BM, 256, 0, stream>>>(x, embh, out);
}
